// GTCNN_v2_83854941487399
// MI455X (gfx1250) — matrix operand order repaired and checked by reading, compiled, not run on hardware
//
#include <hip/hip_runtime.h>
#include <hip/hip_bf16.h>
#include <math.h>

typedef __attribute__((ext_vector_type(16))) _Float16 v16h;
typedef __attribute__((ext_vector_type(8)))  float    v8f;

#define T_DIM 24
#define S_DIM 200
#define N_NODES 4800          // T_DIM * S_DIM
#define C_IN 8
#define C_W 32
#define KORD 3
#define NB 2

#define S_PADM 208            // S_DIM padded to /16 (A rows of As GEMM)
#define S_PADK 224            // S_DIM padded to /32 (K of As GEMM)
#define T_PAD  32             // T_DIM padded to /16 and /32
#define PSLICE (S_DIM * C_W)  // 6400: one t-slice of a [T,S,C] tensor
#define QSLICE (S_PADM * C_W) // 6656: one t-slice of padded Q
#define NSC    (S_DIM * C_W)  // N of the T-GEMM (6400)
#define F_W    64             // FEAT width (40 padded to /32)

// ---------------------------------------------------------------------------
// Batched WMMA GEMM, tile-exact contract:
//   grid = (N/32, M/16, batch); caller guarantees:
//   - A readable [M x K] (row-major, lda=K), B readable [K x N], C writable [M x N]
//   - M%16==0, N%32==0, K%32==0 (zero padding supplies the slack)
// Each wave computes a 16x32 C tile: one A fragment feeds two WMMAs per
// 32-wide k-step. Loads for a step are all issued before the converts and
// drain via staggered partial s_wait_loadcnt.
// ---------------------------------------------------------------------------
__global__ void __launch_bounds__(32, 1)
wmma_gemm_kernel(const float* __restrict__ A,
                 const float* __restrict__ B,
                 float* __restrict__ C,
                 const float* __restrict__ bias,
                 int N, int K,
                 int sA, int sB, int sC,
                 int accumulate, int relu_act)
{
    const int lane = threadIdx.x & 31;
    const int hi   = lane >> 4;      // 0: lanes 0-15, 1: lanes 16-31
    const int lm   = lane & 15;
    const int bn   = blockIdx.x * 32 + lm;   // first column of N-pair
    const int am   = blockIdx.y * 16 + lm;   // A-fragment row
    const int cm0  = blockIdx.y * 16 + 8 * hi;

    A += blockIdx.z * sA;
    B += blockIdx.z * sB;
    C += blockIdx.z * sC;

    v8f acc0 = {}, acc1 = {};
    if (accumulate) {
#pragma unroll
        for (int r = 0; r < 8; ++r) {
            acc0[r] = C[(cm0 + r) * N + bn];
            acc1[r] = C[(cm0 + r) * N + bn + 16];
        }
    }

    for (int k0 = 0; k0 < K; k0 += 32) {
        const float* Arow = A + am * K + k0;
        const float* Bcol = B + (k0 + (hi << 4)) * N + bn;
        float fa[16], fb0[16], fb1[16];
#pragma unroll
        for (int v = 0; v < 8; ++v) {
            // A 16x32 f16 fragment: pair v covers
            //   k = 16*hi + 2*v + {0,1}  (consecutive -> float2)
            float2 f = *(const float2*)(Arow + ((hi << 4) | (v << 1)));
            fa[2 * v]     = f.x;
            fa[2 * v + 1] = f.y;
        }
#pragma unroll
        for (int e = 0; e < 16; ++e) {   // B 32x16 fragments: k = 16*hi + e
            fb0[e] = Bcol[e * N];
            fb1[e] = Bcol[e * N + 16];
        }
        v16h af, bf0, bf1;
#pragma unroll
        for (int e = 0; e < 16; ++e) {
            af[e]  = (_Float16)fa[e];
            bf0[e] = (_Float16)fb0[e];
            bf1[e] = (_Float16)fb1[e];
        }
        acc0 = __builtin_amdgcn_wmma_f32_16x16x32_f16(
            false, af, false, bf0, (short)0, acc0, false, false);
        acc1 = __builtin_amdgcn_wmma_f32_16x16x32_f16(
            false, af, false, bf1, (short)0, acc1, false, false);
    }

    const float bv0 = bias ? bias[bn] : 0.0f;
    const float bv1 = bias ? bias[bn + 16] : 0.0f;
#pragma unroll
    for (int r = 0; r < 8; ++r) {
        float v0 = acc0[r] + bv0;
        float v1 = acc1[r] + bv1;
        if (relu_act) {
            v0 = v0 > 0.0f ? v0 : 0.0f;
            v1 = v1 > 0.0f ? v1 : 0.0f;
        }
        C[(cm0 + r) * N + bn]      = v0;
        C[(cm0 + r) * N + bn + 16] = v1;
    }
}

// ---------------------------------------------------------------------------
// Small helper kernels
// ---------------------------------------------------------------------------
__global__ void rowsum_rsqrt_kernel(const float* __restrict__ A,
                                    float* __restrict__ dinv, int n)
{
    int i = blockIdx.x * blockDim.x + threadIdx.x;
    if (i < n) {
        float s = 0.0f;
        for (int j = 0; j < n; ++j) s += A[i * n + j];
        dinv[i] = rsqrtf(s);
    }
}

// An[DHxDW] = zero-padded D^{-1/2} A D^{-1/2} * 0.5 of A[nxn]
__global__ void sym_norm_pad_kernel(const float* __restrict__ A,
                                    const float* __restrict__ dinv,
                                    float* __restrict__ An,
                                    int n, int DH, int DW)
{
    int idx = blockIdx.x * blockDim.x + threadIdx.x;
    if (idx < DH * DW) {
        int i = idx / DW, j = idx - i * DW;
        An[idx] = (i < n && j < n) ? A[i * n + j] * dinv[i] * dinv[j] * 0.5f : 0.0f;
    }
}

// dst[DHxDW] = zero-padded copy of src[SHxSW]
__global__ void pad_copy_kernel(const float* __restrict__ src,
                                float* __restrict__ dst,
                                int SH, int SW, int DH, int DW)
{
    int idx = blockIdx.x * blockDim.x + threadIdx.x;
    if (idx < DH * DW) {
        int i = idx / DW, j = idx - i * DW;
        dst[idx] = (i < SH && j < SW) ? src[i * SW + j] : 0.0f;
    }
}

__global__ void zero_kernel(float* __restrict__ p, int n)
{
    int i = blockIdx.x * blockDim.x + threadIdx.x;
    if (i < n) p[i] = 0.0f;
}

// U[t][j] = s2*P[t][j] + s3*Q[t][j]   (Q has padded slice stride QSLICE)
__global__ void combine_u_kernel(const float* __restrict__ P,
                                 const float* __restrict__ Q,
                                 float* __restrict__ U,
                                 const float* __restrict__ s, int b)
{
    int j = blockIdx.x * blockDim.x + threadIdx.x;
    int t = blockIdx.y;
    if (j < PSLICE)
        U[t * PSLICE + j] = s[b * 4 + 2] * P[t * PSLICE + j]
                          + s[b * 4 + 3] * Q[t * QSLICE + j];
}

// Pn[t][j] = s0*P[t][j] + s1*Q[t][j] + R[t][j]
__global__ void combine_pnew_kernel(const float* __restrict__ P,
                                    const float* __restrict__ Q,
                                    const float* __restrict__ R,
                                    float* __restrict__ Pn,
                                    const float* __restrict__ s, int b)
{
    int j = blockIdx.x * blockDim.x + threadIdx.x;
    int t = blockIdx.y;
    if (j < PSLICE)
        Pn[t * PSLICE + j] = s[b * 4 + 0] * P[t * PSLICE + j]
                           + s[b * 4 + 1] * Q[t * QSLICE + j]
                           + R[t * PSLICE + j];
}

// feat[node, 0:32]=tanh(acc); [32:40]=x_in; [40:64]=0   (width F_W=64)
__global__ void build_feat_kernel(const float* __restrict__ acc,
                                  const float* __restrict__ x_in,
                                  float* __restrict__ feat, int n)
{
    int i = blockIdx.x * blockDim.x + threadIdx.x;
    if (i < n) {
        int node = i / F_W;
        int c    = i - node * F_W;
        float v = 0.0f;
        if (c < C_W)               v = tanhf(acc[node * C_W + c]);
        else if (c < C_W + C_IN)   v = x_in[node * C_IN + (c - C_W)];
        feat[i] = v;
    }
}

// out[i] = OUTP[i*32] + b_last[0]
__global__ void extract_out_kernel(const float* __restrict__ outp,
                                   const float* __restrict__ b_last,
                                   float* __restrict__ out, int n)
{
    int i = blockIdx.x * blockDim.x + threadIdx.x;
    if (i < n) out[i] = outp[i * 32] + b_last[0];
}

// ---------------------------------------------------------------------------
// Host-side orchestration
// ---------------------------------------------------------------------------
extern "C" void kernel_launch(void* const* d_in, const int* in_sizes, int n_in,
                              void* d_out, int out_size, void* d_ws, size_t ws_size,
                              hipStream_t stream)
{
    (void)in_sizes; (void)n_in; (void)out_size; (void)ws_size;

    const float* x_in    = (const float*)d_in[0];   // [4800, 8]
    const float* Adj_t   = (const float*)d_in[1];   // [24, 24]
    const float* Adj_s   = (const float*)d_in[2];   // [200, 200]
    const float* H       = (const float*)d_in[3];   // [2, 3, 32, 32]
    const float* s_vec   = (const float*)d_in[4];   // [2, 4]
    const float* W_first = (const float*)d_in[5];   // [8, 32]
    const float* b_first = (const float*)d_in[6];   // [32]
    const float* W_last  = (const float*)d_in[7];   // [32, 1]
    const float* b_last  = (const float*)d_in[8];   // [1]
    const float* mlp_W1  = (const float*)d_in[9];   // [2, 40, 32]
    const float* mlp_b1  = (const float*)d_in[10];  // [2, 32]
    const float* mlp_W2  = (const float*)d_in[11];  // [2, 32, 32]
    const float* mlp_b2  = (const float*)d_in[12];  // [2, 32]
    float* out = (float*)d_out;                     // [4800, 1]

    // Workspace carve-up (floats). P-type buffers carry a 24*C_W slack row
    // region (zeroed) so the As-GEMM may read K up to S_PADK on the last t.
    const int PSLACK = (S_PADK - S_DIM) * C_W;     // 768
    float* ws    = (float*)d_ws;
    float* At_n  = ws; ws += T_PAD * T_PAD;        // [32x32] zero-padded
    float* As_n  = ws; ws += S_PADM * S_PADK;      // [208x224] zero-padded
    float* dinvt = ws; ws += 32;
    float* dinvs = ws; ws += 224;
    float* X     = ws; ws += N_NODES * C_W + PSLACK;
    float* Pa    = ws; ws += N_NODES * C_W + PSLACK;
    float* Pb    = ws; ws += N_NODES * C_W + PSLACK;
    float* Q     = ws; ws += T_DIM * QSLICE;       // padded slice stride
    float* U     = ws; ws += T_PAD * NSC;          // rows 24..31 zeroed
    float* R     = ws; ws += T_PAD * NSC;          // rows 24..31 are store slack
    float* ACC   = ws; ws += N_NODES * C_W;
    float* FEAT  = ws; ws += N_NODES * F_W;
    float* HID   = ws; ws += N_NODES * C_W;
    float* XIN   = ws; ws += N_NODES * C_W;        // x_in padded to 32 cols
    float* WF    = ws; ws += C_W * C_W;            // W_first padded [32x32]
    float* W1P   = ws; ws += NB * F_W * C_W;       // mlp_W1 padded [64x32] x2
    float* WL    = ws; ws += C_W * 32;             // W_last padded [32x32]
    float* OUTP  = ws; ws += N_NODES * 32;

    auto gemm = [&](const float* A, const float* B, float* C, const float* bias,
                    int M, int N, int K, int sA, int sB, int sC,
                    int batch, int accum, int relu) {
        dim3 grid(N / 32, M / 16, batch);
        wmma_gemm_kernel<<<grid, 32, 0, stream>>>(A, B, C, bias, N, K,
                                                  sA, sB, sC, accum, relu);
    };
    auto launch1d = [&](int n) { return dim3((n + 255) / 256); };

    // ---- normalization (zero-padded into WMMA-ready layouts) ----
    rowsum_rsqrt_kernel<<<1, 32, 0, stream>>>(Adj_t, dinvt, T_DIM);
    rowsum_rsqrt_kernel<<<1, 256, 0, stream>>>(Adj_s, dinvs, S_DIM);
    sym_norm_pad_kernel<<<launch1d(T_PAD * T_PAD), 256, 0, stream>>>(
        Adj_t, dinvt, At_n, T_DIM, T_PAD, T_PAD);
    sym_norm_pad_kernel<<<launch1d(S_PADM * S_PADK), 256, 0, stream>>>(
        Adj_s, dinvs, As_n, S_DIM, S_PADM, S_PADK);

    // ---- pack inputs into padded layouts; zero read-slack regions ----
    pad_copy_kernel<<<launch1d(N_NODES * C_W), 256, 0, stream>>>(
        x_in, XIN, N_NODES, C_IN, N_NODES, C_W);
    pad_copy_kernel<<<launch1d(C_W * C_W), 256, 0, stream>>>(
        W_first, WF, C_IN, C_W, C_W, C_W);
    pad_copy_kernel<<<launch1d(F_W * C_W), 256, 0, stream>>>(
        mlp_W1, W1P, C_W + C_IN, C_W, F_W, C_W);
    pad_copy_kernel<<<launch1d(F_W * C_W), 256, 0, stream>>>(
        mlp_W1 + (C_W + C_IN) * C_W, W1P + F_W * C_W, C_W + C_IN, C_W, F_W, C_W);
    pad_copy_kernel<<<launch1d(C_W * 32), 256, 0, stream>>>(
        W_last, WL, C_W, 1, C_W, 32);
    zero_kernel<<<launch1d(PSLACK), 256, 0, stream>>>(X + N_NODES * C_W, PSLACK);
    zero_kernel<<<launch1d(PSLACK), 256, 0, stream>>>(Pa + N_NODES * C_W, PSLACK);
    zero_kernel<<<launch1d((T_PAD - T_DIM) * NSC), 256, 0, stream>>>(
        U + T_DIM * NSC, (T_PAD - T_DIM) * NSC);

    // ---- first layer: X = x_in @ W_first + b_first  (padded K=32) ----
    gemm(XIN, WF, X, b_first, N_NODES, C_W, C_W, 0, 0, 0, 1, 0, 0);

    dim3 cgrid((PSLICE + 255) / 256, T_DIM);

    for (int b = 0; b < NB; ++b) {
        const float* Hb = H + b * KORD * C_W * C_W;

        // acc = X @ H[b,0]
        gemm(X, Hb, ACC, nullptr, N_NODES, C_W, C_W, 0, 0, 0, 1, 0, 0);

        const float* pcur = X;
        float* pbufs[2] = { Pa, Pb };
        for (int k = 1; k < KORD; ++k) {
            // Q[t] = As_n @ pcur[t]  (batched over T; M=208,K=224 tile-exact,
            //   pad rows/cols of As_n are zero; pcur slack rows are zero)
            gemm(As_n, pcur, Q, nullptr, S_PADM, C_W, S_PADK,
                 0, PSLICE, QSLICE, T_DIM, 0, 0);
            // U = s2*pcur + s3*Q
            combine_u_kernel<<<cgrid, 256, 0, stream>>>(pcur, Q, U, s_vec, b);
            // R = At_n @ U  as [32 x 6400] (At pad rows zero -> R slack rows zero)
            gemm(At_n, U, R, nullptr, T_PAD, NSC, T_PAD, 0, 0, 0, 1, 0, 0);
            // p_next = s0*pcur + s1*Q + R  ( = Adj @ pcur )
            float* pnext = pbufs[k - 1];
            combine_pnew_kernel<<<cgrid, 256, 0, stream>>>(pcur, Q, R, pnext, s_vec, b);
            // acc += p_next @ H[b,k]
            gemm(pnext, Hb + k * C_W * C_W, ACC, nullptr,
                 N_NODES, C_W, C_W, 0, 0, 0, 1, 1, 0);
            pcur = pnext;
        }

        // feat = [tanh(acc), x_in, 0-pad] width 64
        build_feat_kernel<<<launch1d(N_NODES * F_W), 256, 0, stream>>>(
            ACC, x_in, FEAT, N_NODES * F_W);
        // h = relu(feat @ W1P[b] + b1[b])   K=64 tile-exact
        gemm(FEAT, W1P + b * F_W * C_W, HID, mlp_b1 + b * C_W,
             N_NODES, C_W, F_W, 0, 0, 0, 1, 0, 1);
        // x = h @ W2[b] + b2[b]
        gemm(HID, mlp_W2 + b * C_W * C_W, X, mlp_b2 + b * C_W,
             N_NODES, C_W, C_W, 0, 0, 0, 1, 0, 0);
    }

    // out = X @ W_last (+ b_last in extract)   N padded to 32
    gemm(X, WL, OUTP, nullptr, N_NODES, 32, C_W, 0, 0, 0, 1, 0, 0);
    extract_out_kernel<<<launch1d(N_NODES), 256, 0, stream>>>(OUTP, b_last, out, N_NODES);
}
